// AcousticGenerator_22076131901995
// MI455X (gfx1250) — matrix operand order repaired and checked by reading, not compiled, not run on hardware
//
#include <hip/hip_runtime.h>
#include <cstddef>
#include <cstdint>

// ---------------- model constants ----------------
#define BATCH   16
#define SEQ     512
#define MELLEN  2048
#define DMODEL  384
#define NHEADS  6
#define DHEAD   64
#define FFDIM   1536
#define NMELS   80

typedef _Float16 h8  __attribute__((ext_vector_type(8)));
typedef _Float16 h16 __attribute__((ext_vector_type(16)));
typedef float    f32x8 __attribute__((ext_vector_type(8)));
typedef unsigned int u32x4 __attribute__((ext_vector_type(4)));
typedef int          i32x4 __attribute__((ext_vector_type(4)));
typedef int          i32x8 __attribute__((ext_vector_type(8)));

// attention dynamic-LDS: K[512][72] f16 + Vt[64][520] f16 + S[4][16][516] f32
#define ATTN_LDS (512*72*2 + 64*520*2 + 4*16*516*4)   // 272384 bytes <= 320KB WGP LDS

// ---- TDM availability / toolchain arity detection ----
#if defined(__HIP_DEVICE_COMPILE__) && defined(__gfx1250__) && \
    __has_builtin(__builtin_amdgcn_tensor_load_to_lds) && \
    __has_builtin(__builtin_amdgcn_s_wait_tensorcnt)
  #define USE_TDM 1
  #if __has_include(<hip/amd_detail/amd_gfx1250_TDM.h>)
    #define TDM_6ARG 1        // therock/clang-23 headers -> 6-arg builtin
  #else
    #define TDM_6ARG 0        // ROCm 7.2 / clang-22 -> 5-arg builtin
  #endif
#else
  #define USE_TDM 0
#endif

// Build a 16-half WMMA fragment from two contiguous 8-half (16B) chunks.
__device__ __forceinline__ h16 ldfrag(const _Float16* p0, const _Float16* p1) {
    h8 a = *(const h8*)p0;
    h8 b = *(const h8*)p1;
    return __builtin_shufflevector(a, b, 0,1,2,3,4,5,6,7,8,9,10,11,12,13,14,15);
}

__device__ __forceinline__ f32x8 wmma16(h16 a, h16 b, f32x8 c) {
    return __builtin_amdgcn_wmma_f32_16x16x32_f16(false, a, false, b, (short)0, c, false, false);
}

// ---------------- small utility kernels ----------------
__global__ __launch_bounds__(256) void k_cvt(const float* __restrict__ s,
                                             _Float16* __restrict__ d, int n) {
    int i = blockIdx.x * 256 + threadIdx.x;
    if (i < n) d[i] = (_Float16)s[i];
}

__global__ void k_cumsum(const int* __restrict__ dur, int* __restrict__ cs) {
    int b = threadIdx.x;
    if (b >= BATCH) return;
    const int* d = dur + b * SEQ;
    int* o = cs + b * SEQ;
    int run = 0;
    for (int s = 0; s < SEQ; ++s) { run += d[s]; o[s] = run; }
}

// duration expansion: one wave per (b, mel-frame); searchsorted-right on cumsum
__global__ __launch_bounds__(256) void k_expand(const float* __restrict__ enc,
                                                const int* __restrict__ cs,
                                                float* __restrict__ y32,
                                                _Float16* __restrict__ y16) {
    int wave = threadIdx.x >> 5, lane = threadIdx.x & 31;
    int bm = blockIdx.x * 8 + wave;                 // [0, B*MEL)
    int b = bm >> 11, m = bm & (MELLEN - 1);
    const int* c = cs + b * SEQ;
    int lo = 0, hi = SEQ;
    while (lo < hi) { int mid = (lo + hi) >> 1; if (c[mid] <= m) lo = mid + 1; else hi = mid; }
    int idx = lo < SEQ - 1 ? lo : SEQ - 1;
    bool valid = m < c[SEQ - 1];
    const float* er = enc + ((size_t)b * SEQ + idx) * DMODEL;
    size_t ob = (size_t)bm * DMODEL;
    #pragma unroll
    for (int j = 0; j < 12; ++j) {
        int cc = lane + 32 * j;
        float v = valid ? er[cc] : 0.f;
        y32[ob + cc] = v;
        y16[ob + cc] = (_Float16)v;
    }
}

// residual already folded into x; LayerNorm over 384, one wave per row
__global__ __launch_bounds__(256) void k_ln(const float* __restrict__ x,
                                            const float* __restrict__ g,
                                            const float* __restrict__ bta,
                                            float* __restrict__ y32,
                                            _Float16* __restrict__ y16, int M) {
    int wave = threadIdx.x >> 5, lane = threadIdx.x & 31;
    int row = blockIdx.x * 8 + wave;
    if (row >= M) return;
    const float* xr = x + (size_t)row * DMODEL;
    float v[12]; float s = 0.f;
    #pragma unroll
    for (int j = 0; j < 12; ++j) { v[j] = xr[lane + 32 * j]; s += v[j]; }
    #pragma unroll
    for (int off = 16; off > 0; off >>= 1) s += __shfl_xor(s, off, 32);
    float mean = s * (1.f / DMODEL);
    float q = 0.f;
    #pragma unroll
    for (int j = 0; j < 12; ++j) { float d = v[j] - mean; q += d * d; }
    #pragma unroll
    for (int off = 16; off > 0; off >>= 1) q += __shfl_xor(q, off, 32);
    float rs = rsqrtf(q * (1.f / DMODEL) + 1e-5f);
    size_t ob = (size_t)row * DMODEL;
    #pragma unroll
    for (int j = 0; j < 12; ++j) {
        int c = lane + 32 * j;
        float o = (v[j] - mean) * rs * g[c] + bta[c];
        y32[ob + c] = o;
        y16[ob + c] = (_Float16)o;
    }
}

// x + tile(pre,3)[..., :384] -> f16
__global__ __launch_bounds__(256) void k_tileadd(const float* __restrict__ x,
                                                 const float* __restrict__ p,
                                                 _Float16* __restrict__ y, int n) {
    int i = blockIdx.x * 256 + threadIdx.x;
    if (i >= n) return;
    int m = i / DMODEL, c = i - m * DMODEL;
    y[i] = (_Float16)(x[i] + p[m * 128 + (c & 127)]);
}

// ---------------- WMMA GEMM: C = act(A[MxK]@W[KxN] + bias (+dur-term) (+resid)) ----------------
// Block tile 128x64, 8 waves, each wave a 16x64 strip (4 accumulators), K-step 32.
__global__ __launch_bounds__(256) void k_gemm(
    const _Float16* __restrict__ A, const _Float16* __restrict__ W,
    const float* __restrict__ bias, const float* __restrict__ resid,
    const int* __restrict__ dur, const float* __restrict__ durw, const float* __restrict__ durb,
    float* __restrict__ out32, _Float16* __restrict__ out16,
    int M, int N, int K, int act)
{
    __shared__ _Float16 As[128][40];   // padded: 20-dword row stride -> conflict-free A frags
    __shared__ _Float16 Wt[64][40];    // W tile transposed: Wt[n][k]

    const int t = threadIdx.x;
    const int wave = t >> 5, lane = t & 31;
    const int ln = lane & 15, hi = lane >> 4;
    const int m0 = blockIdx.y * 128, n0 = blockIdx.x * 64;

    const f32x8 z = {0.f,0.f,0.f,0.f,0.f,0.f,0.f,0.f};
    f32x8 acc[4] = {z, z, z, z};

    const int arow = t >> 1, akoff = (t & 1) * 16;
    const int wrow = t >> 3, wnoff = (t & 7) * 8;

    for (int k0 = 0; k0 < K; k0 += 32) {
        __syncthreads();
        { // A tile: 128x32 halves, vectorized; prefetch next K-step tile
            const h8* src = (const h8*)(A + (size_t)(m0 + arow) * K + k0 + akoff);
            *(h8*)(&As[arow][akoff])     = src[0];
            *(h8*)(&As[arow][akoff + 8]) = src[1];
            if (k0 + 32 < K)
                __builtin_prefetch(A + (size_t)(m0 + arow) * K + k0 + 32 + akoff, 0, 0);
        }
        { // W tile: 32x64 -> transposed into Wt (zero-pad when n >= N)
            if (n0 + wnoff < N) {
                h8 wv = *(const h8*)(W + (size_t)(k0 + wrow) * N + n0 + wnoff);
                #pragma unroll
                for (int j = 0; j < 8; ++j) Wt[wnoff + j][wrow] = wv[j];
            } else {
                #pragma unroll
                for (int j = 0; j < 8; ++j) Wt[wnoff + j][wrow] = (_Float16)0.f;
            }
        }
        __syncthreads();

        const _Float16* ap = &As[wave * 16 + ln][0];
        h16 af = ldfrag(ap + 8 * hi, ap + 16 + 8 * hi);     // A layout: K=(h&7)+8hi+16(h>>3)
        #pragma unroll
        for (int nt = 0; nt < 4; ++nt) {
            const _Float16* bp = &Wt[nt * 16 + ln][0];
            h16 bf = ldfrag(bp + 8 * hi, bp + 16 + 8 * hi); // B layout: K=(h&7)+8hi+16(h>>3)
            acc[nt] = wmma16(af, bf, acc[nt]);
        }
    }

    #pragma unroll
    for (int nt = 0; nt < 4; ++nt) {
        const int n = n0 + nt * 16 + ln;
        if (n < N) {
            #pragma unroll
            for (int r = 0; r < 8; ++r) {
                const int m = m0 + wave * 16 + 8 * hi + r;
                float vv = acc[nt][r];
                if (bias)  vv += bias[n];
                if (dur)   vv += (float)dur[m] * durw[n] + durb[n];
                if (resid) vv += resid[(size_t)m * N + n];
                if (act == 1)      vv = fmaxf(vv, 0.f);
                else if (act == 2) vv = (vv > 0.f) ? vv : 0.1f * vv;
                if (out32) out32[(size_t)m * N + n] = vv;
                if (out16) out16[(size_t)m * N + n] = (_Float16)vv;
            }
        }
    }
}

// ---------------- fused attention: softmax(Q Kt / 8) V, Tk=512, dh=64 ----------------
// grid(Tq/64, H, B), 128 threads (4 waves); each wave owns a 16-row q tile.
// K tile staged by the Tensor Data Mover (TDM) with native LDS row padding
// (72-half stride); V transposed manually; exact softmax over the full 512 row.
__global__ __launch_bounds__(128) void k_attn(
    const _Float16* __restrict__ Q, const _Float16* __restrict__ Kmat,
    const _Float16* __restrict__ Vmat, _Float16* __restrict__ O, int Tq)
{
    extern __shared__ __align__(16) char smem[];
    _Float16* Ks = (_Float16*)smem;                         // [512][72]
    _Float16* Vt = (_Float16*)(smem + 512 * 72 * 2);        // [64][520] (transposed)
    float*    S  = (float*)(smem + 512 * 72 * 2 + 64 * 520 * 2); // [4][16][516]

    const int t = threadIdx.x;
    const int wave = t >> 5, lane = t & 31;
    const int ln = lane & 15, hi = lane >> 4;
    const int bz = blockIdx.z, hy = blockIdx.y;
    const int q0 = blockIdx.x * 64 + wave * 16;

    const _Float16* Kg = Kmat + (size_t)bz * SEQ * DMODEL + hy * DHEAD;
    const _Float16* Vg = Vmat + (size_t)bz * SEQ * DMODEL + hy * DHEAD;

#if USE_TDM
    if (wave == 0) {
        // D# group0: count=1 | lds_addr | 57-bit global addr | type=2 ("image")
        uint64_t ga = (uint64_t)(uintptr_t)Kg;
        uint32_t lds_addr = (uint32_t)(uintptr_t)Ks;   // low 32 bits of LDS aperture = LDS offset
        u32x4 g0;
        g0[0] = 1u;
        g0[1] = lds_addr;
        g0[2] = (uint32_t)ga;
        g0[3] = ((uint32_t)(ga >> 32) & 0x01FFFFFFu) | (2u << 30);
        // D# group1: data_size=2B, pad_enable, pad_interval=32 DW (one 64-half row),
        // pad_amount=4 DW (8 halves) -> 72-half row stride in LDS.
        // tensor_dim0=384, tensor_dim1=512, tile_dim0=64, tile_dim1=512, stride0=384.
        i32x8 g1;
        g1[0] = (1 << 16) | (1 << 20) | (4 << 22) | (3 << 25);
        g1[1] = (int)(384u << 16);     // tensor_dim0[15:0] at bits [63:48]
        g1[2] = (int)(512u << 16);     // tensor_dim0 hi=0 | tensor_dim1[15:0]
        g1[3] = (int)(64u << 16);      // tensor_dim1 hi=0 | tile_dim0
        g1[4] = 512;                   // tile_dim1 (tile_dim2=0)
        g1[5] = 384;                   // tensor_dim0_stride low32
        g1[6] = 0;
        g1[7] = 0;
        i32x4 gz = {0, 0, 0, 0};
  #if TDM_6ARG
        i32x8 gz8 = {0, 0, 0, 0, 0, 0, 0, 0};
        __builtin_amdgcn_tensor_load_to_lds(g0, g1, gz, gz, gz8, 0);
  #else
        __builtin_amdgcn_tensor_load_to_lds(g0, g1, gz, gz, 0);
  #endif
        __builtin_amdgcn_s_wait_tensorcnt(0);
    }
    for (int i = t; i < SEQ * 8; i += 128) {       // V only: transpose while loading
        int row = i >> 3, d8 = (i & 7) * 8;
        h8 vv = *(const h8*)(Vg + (size_t)row * DMODEL + d8);
        #pragma unroll
        for (int j = 0; j < 8; ++j) Vt[(d8 + j) * 520 + row] = vv[j];
    }
#else
    for (int i = t; i < SEQ * 8; i += 128) {
        int row = i >> 3, d8 = (i & 7) * 8;
        h8 kv = *(const h8*)(Kg + (size_t)row * DMODEL + d8);
        *(h8*)(&Ks[row * 72 + d8]) = kv;
        h8 vv = *(const h8*)(Vg + (size_t)row * DMODEL + d8);
        #pragma unroll
        for (int j = 0; j < 8; ++j) Vt[(d8 + j) * 520 + row] = vv[j];
    }
#endif
    __syncthreads();

    // Q fragments straight from global (A layout)
    const _Float16* Qg = Q + (size_t)(bz * Tq + q0 + ln) * DMODEL + hy * DHEAD;
    h16 qa0 = ldfrag(Qg + 8 * hi,      Qg + 16 + 8 * hi);
    h16 qa1 = ldfrag(Qg + 32 + 8 * hi, Qg + 48 + 8 * hi);

    float* Sw = S + wave * 16 * 516;
    const f32x8 z = {0.f,0.f,0.f,0.f,0.f,0.f,0.f,0.f};

    // scores = Q @ K^T * 0.125  (B operand reads K rows contiguously: B[k][n]=K[n-col][d])
    for (int kc = 0; kc < 32; ++kc) {
        const _Float16* kp = &Ks[(kc * 16 + ln) * 72];
        h16 b0 = ldfrag(kp + 8 * hi,      kp + 16 + 8 * hi);
        h16 b1 = ldfrag(kp + 32 + 8 * hi, kp + 48 + 8 * hi);
        f32x8 sc = z;
        sc = wmma16(qa0, b0, sc);
        sc = wmma16(qa1, b1, sc);
        #pragma unroll
        for (int r = 0; r < 8; ++r)
            Sw[(r + 8 * hi) * 516 + kc * 16 + ln] = sc[r] * 0.125f;
    }

    // exact softmax per row (wave-local; LDS ops of one wave are in order)
    for (int r = 0; r < 16; ++r) {
        float* row = Sw + r * 516;
        float mx = -3.0e38f;
        #pragma unroll
        for (int j = 0; j < 16; ++j) mx = fmaxf(mx, row[lane + 32 * j]);
        #pragma unroll
        for (int off = 16; off > 0; off >>= 1) mx = fmaxf(mx, __shfl_xor(mx, off, 32));
        float sum = 0.f;
        #pragma unroll
        for (int j = 0; j < 16; ++j) {
            float e = __expf(row[lane + 32 * j] - mx);
            row[lane + 32 * j] = e;
            sum += e;
        }
        #pragma unroll
        for (int off = 16; off > 0; off >>= 1) sum += __shfl_xor(sum, off, 32);
        float inv = 1.f / sum;
        _Float16* pr = (_Float16*)row;       // pack P as f16 in place (reads precede writes)
        #pragma unroll
        for (int j = 0; j < 16; ++j) pr[lane + 32 * j] = (_Float16)(row[lane + 32 * j] * inv);
    }

    // O = P @ V
    f32x8 oacc[4] = {z, z, z, z};
    const _Float16* Pp = (const _Float16*)(Sw + (size_t)ln * 516);
    for (int kk = 0; kk < SEQ; kk += 32) {
        h16 pa = ldfrag(Pp + kk + 8 * hi, Pp + kk + 16 + 8 * hi);
        #pragma unroll
        for (int nt = 0; nt < 4; ++nt) {
            const _Float16* vp = &Vt[(nt * 16 + ln) * 520];
            h16 vb = ldfrag(vp + kk + 8 * hi, vp + kk + 16 + 8 * hi);
            oacc[nt] = wmma16(pa, vb, oacc[nt]);
        }
    }

    #pragma unroll
    for (int nt = 0; nt < 4; ++nt)
        #pragma unroll
        for (int r = 0; r < 8; ++r) {
            int m = q0 + 8 * hi + r;
            O[(size_t)(bz * Tq + m) * DMODEL + hy * DHEAD + nt * 16 + ln] = (_Float16)oacc[nt][r];
        }
}

// ---------------- host orchestration ----------------
extern "C" void kernel_launch(void* const* d_in, const int* in_sizes, int n_in,
                              void* d_out, int out_size, void* d_ws, size_t ws_size,
                              hipStream_t stream)
{
    (void)n_in; (void)out_size; (void)ws_size;
    const int P0 = 5;                 // first params leaf (after 5 tensor inputs)
    const int M1 = BATCH * SEQ;       // 8192
    const int M2 = BATCH * MELLEN;    // 32768

    char* wsb = (char*)d_ws;
    size_t cur = 0;
    auto balloc = [&](size_t bytes) { size_t r = cur; cur = (cur + bytes + 255) & ~(size_t)255; return r; };

    auto cvtW = [&](int leaf) -> _Float16* {
        size_t n = (size_t)in_sizes[leaf];
        _Float16* dst = (_Float16*)(wsb + balloc(n * sizeof(_Float16)));
        k_cvt<<<dim3((unsigned)((n + 255) / 256)), dim3(256), 0, stream>>>((const float*)d_in[leaf], dst, (int)n);
        return dst;
    };

    struct LayerW {
        _Float16 *q, *k, *v, *o, *f1, *f2;
        const float *qb, *kb, *vb, *ob, *f1b, *f2b, *ln1g, *ln1b, *ln2g, *ln2b;
    };
    // pytree leaf order within a layer (sorted keys, 'b' before 'w'):
    // 0 k.b 1 k.w 2 o.b 3 o.w 4 q.b 5 q.w 6 v.b 7 v.w 8 ffn1.b 9 ffn1.w
    // 10 ffn2.b 11 ffn2.w 12 ln1b 13 ln1g 14 ln2b 15 ln2g
    auto mkLayer = [&](int base) {
        LayerW L;
        L.k = cvtW(base + 1);  L.o = cvtW(base + 3);
        L.q = cvtW(base + 5);  L.v = cvtW(base + 7);
        L.f1 = cvtW(base + 9); L.f2 = cvtW(base + 11);
        L.kb  = (const float*)d_in[base + 0];  L.ob  = (const float*)d_in[base + 2];
        L.qb  = (const float*)d_in[base + 4];  L.vb  = (const float*)d_in[base + 6];
        L.f1b = (const float*)d_in[base + 8];  L.f2b = (const float*)d_in[base + 10];
        L.ln1b = (const float*)d_in[base + 12]; L.ln1g = (const float*)d_in[base + 13];
        L.ln2b = (const float*)d_in[base + 14]; L.ln2g = (const float*)d_in[base + 15];
        return L;
    };

    // params sorted keys: cross(2x16), dur(2), enc(4x16), f01, f02, input, lang1,
    // lang2, out, pre1, pre2, pre3, spk1, spk2   (spk/lang/f0 paths are dead code)
    LayerW crossL[2];
    for (int i = 0; i < 2; ++i) crossL[i] = mkLayer(P0 + 16 * i);
    const float* durB = (const float*)d_in[P0 + 32];
    const float* durW = (const float*)d_in[P0 + 33];
    LayerW encL[4];
    for (int i = 0; i < 4; ++i) encL[i] = mkLayer(P0 + 34 + 16 * i);
    const float* inputB = (const float*)d_in[P0 + 102]; _Float16* inputW = cvtW(P0 + 103);
    const float* outB   = (const float*)d_in[P0 + 108]; _Float16* outW   = cvtW(P0 + 109);
    const float* pre1B  = (const float*)d_in[P0 + 110]; _Float16* pre1W  = cvtW(P0 + 111);
    const float* pre2B  = (const float*)d_in[P0 + 112]; _Float16* pre2W  = cvtW(P0 + 113);
    const float* pre3B  = (const float*)d_in[P0 + 114]; _Float16* pre3W  = cvtW(P0 + 115);

    // activation buffers
    float*    X32A = (float*)(wsb + balloc((size_t)M1 * DMODEL * 4));
    float*    X32B = (float*)(wsb + balloc((size_t)M2 * DMODEL * 4));
    float*    X32C = (float*)(wsb + balloc((size_t)M2 * DMODEL * 4));
    _Float16* X16A = (_Float16*)(wsb + balloc((size_t)M2 * DMODEL * 2));
    _Float16* X16B = (_Float16*)(wsb + balloc((size_t)M2 * DMODEL * 2));
    _Float16* X16C = (_Float16*)(wsb + balloc((size_t)M2 * DMODEL * 2));
    _Float16* Q16  = (_Float16*)(wsb + balloc((size_t)M2 * DMODEL * 2));
    _Float16* K16  = (_Float16*)(wsb + balloc((size_t)M1 * DMODEL * 2));
    _Float16* V16  = (_Float16*)(wsb + balloc((size_t)M1 * DMODEL * 2));
    _Float16* E16  = (_Float16*)(wsb + balloc((size_t)M1 * DMODEL * 2));
    _Float16* H16  = (_Float16*)(wsb + balloc((size_t)M2 * FFDIM * 2));
    _Float16* P216 = (_Float16*)(wsb + balloc((size_t)M2 * 256 * 2));
    float*    P332 = (float*)(wsb + balloc((size_t)M2 * 128 * 4));
    int*      CS   = (int*)(wsb + balloc((size_t)BATCH * SEQ * 4));

    const int* dur = (const int*)d_in[1];

    (void)hipFuncSetAttribute(reinterpret_cast<const void*>(&k_attn),
                              hipFuncAttributeMaxDynamicSharedMemorySize, ATTN_LDS);

    auto gemm = [&](const _Float16* A, const _Float16* W, const float* bias, const float* resid,
                    const int* dr, const float* drw, const float* drb,
                    float* o32, _Float16* o16, int M, int N, int K, int act) {
        dim3 g((unsigned)((N + 63) / 64), (unsigned)(M / 128));
        k_gemm<<<g, 256, 0, stream>>>(A, W, bias, resid, dr, drw, drb, o32, o16, M, N, K, act);
    };
    auto lnorm = [&](const float* x, const float* g, const float* b, float* y32, _Float16* y16, int M) {
        k_ln<<<dim3((unsigned)(M / 8)), 256, 0, stream>>>(x, g, b, y32, y16, M);
    };
    auto attn = [&](const _Float16* Qp, const _Float16* Kp, const _Float16* Vp, _Float16* Op, int Tq) {
        dim3 g((unsigned)(Tq / 64), NHEADS, BATCH);
        k_attn<<<g, 128, ATTN_LDS, stream>>>(Qp, Kp, Vp, Op, Tq);
    };

    // phoneme_emb -> f16
    {   size_t n = (size_t)M1 * DMODEL;
        k_cvt<<<dim3((unsigned)((n + 255) / 256)), 256, 0, stream>>>((const float*)d_in[0], X16A, (int)n); }

    // input proj + fused duration embedding
    gemm(X16A, inputW, inputB, nullptr, dur, durW, durB, X32A, X16B, M1, DMODEL, DMODEL, 0);

    // encoder (self-attention), invariant: x f32 in X32A, x f16 in X16B
    for (int i = 0; i < 4; ++i) {
        const LayerW& L = encL[i];
        gemm(X16B, L.q, L.qb, nullptr, nullptr, nullptr, nullptr, nullptr, Q16, M1, DMODEL, DMODEL, 0);
        gemm(X16B, L.k, L.kb, nullptr, nullptr, nullptr, nullptr, nullptr, K16, M1, DMODEL, DMODEL, 0);
        gemm(X16B, L.v, L.vb, nullptr, nullptr, nullptr, nullptr, nullptr, V16, M1, DMODEL, DMODEL, 0);
        attn(Q16, K16, V16, X16C, SEQ);
        gemm(X16C, L.o, L.ob, X32A, nullptr, nullptr, nullptr, X32B, nullptr, M1, DMODEL, DMODEL, 0);
        lnorm(X32B, L.ln1g, L.ln1b, X32A, X16B, M1);
        gemm(X16B, L.f1, L.f1b, nullptr, nullptr, nullptr, nullptr, nullptr, H16, M1, FFDIM, DMODEL, 1);
        gemm(H16, L.f2, L.f2b, X32A, nullptr, nullptr, nullptr, X32B, nullptr, M1, DMODEL, FFDIM, 0);
        lnorm(X32B, L.ln2g, L.ln2b, X32A, X16B, M1);
    }

    // keep encoder output (f16) for cross-attn K/V; expand by durations
    {   size_t n = (size_t)M1 * DMODEL;
        k_cvt<<<dim3((unsigned)((n + 255) / 256)), 256, 0, stream>>>(X32A, E16, (int)n); }
    k_cumsum<<<dim3(1), dim3(32), 0, stream>>>(dur, CS);
    k_expand<<<dim3((unsigned)(M2 / 8)), 256, 0, stream>>>(X32A, CS, X32B, X16B);

    // cross blocks (conditioning path is dead in the reference), x f32=X32B, f16=X16B
    for (int i = 0; i < 2; ++i) {
        const LayerW& L = crossL[i];
        gemm(X16B, L.q, L.qb, nullptr, nullptr, nullptr, nullptr, nullptr, Q16, M2, DMODEL, DMODEL, 0);
        gemm(E16,  L.k, L.kb, nullptr, nullptr, nullptr, nullptr, nullptr, K16, M1, DMODEL, DMODEL, 0);
        gemm(E16,  L.v, L.vb, nullptr, nullptr, nullptr, nullptr, nullptr, V16, M1, DMODEL, DMODEL, 0);
        attn(Q16, K16, V16, X16C, MELLEN);
        gemm(X16C, L.o, L.ob, X32B, nullptr, nullptr, nullptr, X32C, nullptr, M2, DMODEL, DMODEL, 0);
        lnorm(X32C, L.ln1g, L.ln1b, X32B, X16B, M2);
        gemm(X16B, L.f1, L.f1b, nullptr, nullptr, nullptr, nullptr, nullptr, H16, M2, FFDIM, DMODEL, 1);
        gemm(H16, L.f2, L.f2b, X32B, nullptr, nullptr, nullptr, X32C, nullptr, M2, DMODEL, FFDIM, 0);
        lnorm(X32C, L.ln2g, L.ln2b, X32B, X16B, M2);
    }

    // pre-net + tiled residual + output projection (straight into d_out)
    gemm(X16B, pre1W, pre1B, nullptr, nullptr, nullptr, nullptr, nullptr, X16C, M2, DMODEL, DMODEL, 2);
    gemm(X16C, pre2W, pre2B, nullptr, nullptr, nullptr, nullptr, nullptr, P216, M2, 256, DMODEL, 2);
    gemm(P216, pre3W, pre3B, nullptr, nullptr, nullptr, nullptr, P332, nullptr, M2, 128, 256, 0);
    {   int n = M2 * DMODEL;
        k_tileadd<<<dim3((unsigned)((n + 255) / 256)), 256, 0, stream>>>(X32B, P332, X16A, n); }
    gemm(X16A, outW, outB, nullptr, nullptr, nullptr, nullptr, (float*)d_out, nullptr, M2, NMELS, DMODEL, 0);
}
